// prop4_87093346828662
// MI455X (gfx1250) — matrix operand order repaired and checked by reading, not compiled, not run on hardware
//
#include <hip/hip_runtime.h>

typedef __attribute__((ext_vector_type(16))) _Float16 v16h;
typedef __attribute__((ext_vector_type(8)))  float    v8f;

#define NB 8
#define NC 16
#define NN 1024
#define NS 12
#define ALPHA 0.05f

// ---------------------------------------------------------------------------
// One propagation step:
//   h_out[b,c,v,s] = ALPHA*x + (1-ALPHA) * (sum_n (adj[b,v,n,s]+I) * h_in[b,c,n,s]) / d[b,v,s]
// d factors out of the n-sum, so we accumulate raw (adj+I) products with
// v_wmma_f32_16x16x32_f16 and apply 1/d in the epilogue; the row degree is
// accumulated for free while staging adj tiles (no separate degree pass).
// grid = (64 v-tiles, 8 batches), block = 128 = 4 wave32; wave w owns s=3w..3w+2.
// Per chunk: stage [16 v][32 n][12 s] of adj and [16 c][32 n][12 s] of h into
// LDS as f16, then 3 WMMAs per wave (M=c, N=v, K=n-chunk).
// ---------------------------------------------------------------------------
__global__ __launch_bounds__(128) void prop_kernel(
    const float* __restrict__ hin,
    const float* __restrict__ x,
    const float* __restrict__ adj,
    float* __restrict__ hout)
{
    const int t    = threadIdx.x;
    const int lane = t & 31;
    const int wid  = t >> 5;       // 0..3
    const int col  = lane & 15;    // N index (target v) for B/D, M index (c) for A
    const int hi   = lane >> 4;    // lane half
    const int v0   = blockIdx.x * 16;
    const int b    = blockIdx.y;

    // LDS tiles (f16): A = h_in [s][c][k32], B = (adj+I)^T [s][v][k32]
    __shared__ _Float16 Ah[NS * 16 * 32];
    __shared__ _Float16 Bh[NS * 16 * 32];
    __shared__ float    dred[128 * NS];
    __shared__ float    dS[16 * NS];

    // loader mapping: thread owns row ivL (v for adj, c for h), 12 float4 per tile
    const int ivL = t >> 3;        // 0..15
    const int j8  = t & 7;         // 0..7

    const float4* adj4 = (const float4*)adj;
    const float4* h4   = (const float4*)hin;
    const size_t adjBase = ((size_t)b * NN + v0) * (size_t)(NN * 3); // float4 units
    const size_t hBase   = ((size_t)b * NC) * (size_t)(NN * 3);

    float4 dpart[3];
    dpart[0] = make_float4(0.f, 0.f, 0.f, 0.f);
    dpart[1] = make_float4(0.f, 0.f, 0.f, 0.f);
    dpart[2] = make_float4(0.f, 0.f, 0.f, 0.f);

    v8f acc0 = {}, acc1 = {}, acc2 = {};

    for (int nc = 0; nc < 32; ++nc) {
        const int n0 = nc * 32;

        // ---- stage adj tile (add self-loop, accumulate row degree) ----
        #pragma unroll
        for (int rep = 0; rep < 12; ++rep) {
            const int in = 4 * j8 + rep / 3;   // source node within chunk
            const int sq = rep % 3;            // s-quad (s = 4*sq .. 4*sq+3)
            float4 q = adj4[adjBase + (size_t)ivL * (NN*3) + (size_t)(n0 + in) * 3 + sq];
            dpart[sq].x += q.x; dpart[sq].y += q.y;
            dpart[sq].z += q.z; dpart[sq].w += q.w;
            const float eye = (v0 + ivL == n0 + in) ? 1.0f : 0.0f;
            q.x += eye; q.y += eye; q.z += eye; q.w += eye;
            Bh[((4*sq + 0) * 16 + ivL) * 32 + in] = (_Float16)q.x;
            Bh[((4*sq + 1) * 16 + ivL) * 32 + in] = (_Float16)q.y;
            Bh[((4*sq + 2) * 16 + ivL) * 32 + in] = (_Float16)q.z;
            Bh[((4*sq + 3) * 16 + ivL) * 32 + in] = (_Float16)q.w;
        }
        // ---- stage h tile ----
        #pragma unroll
        for (int rep = 0; rep < 12; ++rep) {
            const int in = 4 * j8 + rep / 3;
            const int sq = rep % 3;
            const float4 q = h4[hBase + (size_t)ivL * (NN*3) + (size_t)(n0 + in) * 3 + sq];
            Ah[((4*sq + 0) * 16 + ivL) * 32 + in] = (_Float16)q.x;
            Ah[((4*sq + 1) * 16 + ivL) * 32 + in] = (_Float16)q.y;
            Ah[((4*sq + 2) * 16 + ivL) * 32 + in] = (_Float16)q.z;
            Ah[((4*sq + 3) * 16 + ivL) * 32 + in] = (_Float16)q.w;
        }
        if (nc + 1 < 32) {  // pull next adj tile toward the caches
            __builtin_prefetch((const void*)&adj4[adjBase + (size_t)ivL * (NN*3)
                                   + (size_t)(n0 + 32 + 4*j8) * 3], 0, 1);
        }
        __syncthreads();

        // ---- one v_wmma_f32_16x16x32_f16 per s (3 per wave) ----
        auto mm = [&](int s, v8f acc) -> v8f {
            v16h af, bf;
            // A 16x32 f16 layout: lane (M=col, half hi): e<8 -> K=8*hi+e ; e>=8 -> K=16+8*hi+(e-8)
            const _Float16* Ar = &Ah[(s * 16 + col) * 32];
            #pragma unroll
            for (int e = 0; e < 8; ++e) af[e] = Ar[8*hi + e];
            #pragma unroll
            for (int e = 0; e < 8; ++e) af[8 + e] = Ar[16 + 8*hi + e];
            // B 32x16 f16 layout: lane (N=col, half hi): e<8 -> K=8*hi+e ; e>=8 -> K=16+8*hi+(e-8)
            const _Float16* Br = &Bh[(s * 16 + col) * 32];
            #pragma unroll
            for (int e = 0; e < 8; ++e) bf[e] = Br[8*hi + e];
            #pragma unroll
            for (int e = 0; e < 8; ++e) bf[8 + e] = Br[16 + 8*hi + e];
            return __builtin_amdgcn_wmma_f32_16x16x32_f16(
                false, af, false, bf, (short)0, acc, false, false);
        };
        acc0 = mm(wid * 3 + 0, acc0);
        acc1 = mm(wid * 3 + 1, acc1);
        acc2 = mm(wid * 3 + 2, acc2);
        __syncthreads();
    }

    // ---- reduce per-thread degree partials -> dS[v][s] = 1/(1 + sum_n adj) ----
    #pragma unroll
    for (int sq = 0; sq < 3; ++sq) {
        dred[t * NS + 4*sq + 0] = dpart[sq].x;
        dred[t * NS + 4*sq + 1] = dpart[sq].y;
        dred[t * NS + 4*sq + 2] = dpart[sq].z;
        dred[t * NS + 4*sq + 3] = dpart[sq].w;
    }
    __syncthreads();
    if (t < 16) {
        #pragma unroll
        for (int s = 0; s < NS; ++s) {
            float d = 1.0f;  // self loop
            #pragma unroll
            for (int j = 0; j < 8; ++j) d += dred[(t * 8 + j) * NS + s];
            dS[t * NS + s] = 1.0f / d;
        }
    }
    __syncthreads();

    // ---- epilogue: D element e holds (M = 8*hi + e, N = col) ----
    auto wb = [&](int s, const v8f acc) {
        const float dv = dS[col * NS + s];
        #pragma unroll
        for (int e = 0; e < 8; ++e) {
            const int c = 8 * hi + e;
            const size_t idx = ((size_t)(b * NC + c) * NN + (size_t)(v0 + col)) * NS + s;
            hout[idx] = ALPHA * x[idx] + (1.0f - ALPHA) * dv * acc[e];
        }
    };
    wb(wid * 3 + 0, acc0);
    wb(wid * 3 + 1, acc1);
    wb(wid * 3 + 2, acc2);
}

// ---------------------------------------------------------------------------
// 1x1 conv / channel mix: out[b,o,n,s] = bias[o] + sum_c W[o,c]*h[b,c,n,s]
// Tiny (6.3 MB in / 6.3 MB out); one thread per (b,n,s) column.
// ---------------------------------------------------------------------------
__global__ __launch_bounds__(256) void mix_kernel(
    const float* __restrict__ h,
    const float* __restrict__ W,
    const float* __restrict__ bias,
    float* __restrict__ out)
{
    __shared__ float Ws[NC * NC];
    __shared__ float bs[NC];
    const int t = threadIdx.x;
    if (t < NC * NC) Ws[t] = W[t];
    if (t < NC) bs[t] = bias[t];
    __syncthreads();

    const size_t colTot = (size_t)NB * NN * NS;
    const size_t colIdx = (size_t)blockIdx.x * blockDim.x + t;
    if (colIdx >= colTot) return;
    const int b   = (int)(colIdx / (NN * NS));
    const int rem = (int)(colIdx % (NN * NS));

    float hv[NC];
    #pragma unroll
    for (int c = 0; c < NC; ++c)
        hv[c] = h[(size_t)(b * NC + c) * (NN * NS) + rem];

    #pragma unroll
    for (int o = 0; o < NC; ++o) {
        float a = bs[o];
        #pragma unroll
        for (int c = 0; c < NC; ++c) a += Ws[o * NC + c] * hv[c];
        out[(size_t)(b * NC + o) * (NN * NS) + rem] = a;
    }
}

extern "C" void kernel_launch(void* const* d_in, const int* in_sizes, int n_in,
                              void* d_out, int out_size, void* d_ws, size_t ws_size,
                              hipStream_t stream) {
    (void)in_sizes; (void)n_in; (void)out_size; (void)ws_size;
    const float* x    = (const float*)d_in[0];
    const float* adj  = (const float*)d_in[1];
    const float* W    = (const float*)d_in[2];
    const float* bias = (const float*)d_in[3];
    float* out = (float*)d_out;

    // workspace: two ping-pong h buffers (8*16*1024*12 fp32 = 6,291,456 B each)
    char* ws = (char*)d_ws;
    float* h0 = (float*)ws;
    float* h1 = (float*)(ws + (size_t)NB * NC * NN * NS * sizeof(float));

    dim3 pg(NN / 16, NB);   // 64 v-tiles x 8 batches
    prop_kernel<<<pg, 128, 0, stream>>>(x,  x, adj, h0);   // h1 = f(x)
    prop_kernel<<<pg, 128, 0, stream>>>(h0, x, adj, h1);   // h2 = f(h1)
    prop_kernel<<<pg, 128, 0, stream>>>(h1, x, adj, h0);   // h3 = f(h2)

    const int cols = NB * NN * NS;
    mix_kernel<<<(cols + 255) / 256, 256, 0, stream>>>(h0, W, bias, out);
}
